// GrammarTreeDecoder_56762287784431
// MI455X (gfx1250) — matrix operand order repaired and checked by reading, not compiled, not run on hardware
//
#include <hip/hip_runtime.h>
#include <hip/hip_bf16.h>

// ---------------------------------------------------------------------------
// GrammarTreeDecoder for MI455X (gfx1250, wave32, WMMA 16x16x32 f16/f32-acc)
//
// Strategy: route tokens by category / child-index, pre-convert weights to
// f16 in workspace (L2-resident), then do gathered 16-token-tile GEMMs with
// v_wmma_f32_16x16x32_f16, fusing the LSTM pointwise math into the GEMM
// epilogue (i/f/g/o tiles accumulated together per wave).
// ---------------------------------------------------------------------------

typedef __attribute__((ext_vector_type(16))) _Float16 v16h;
typedef __attribute__((ext_vector_type(8)))  _Float16 v8h;
typedef __attribute__((ext_vector_type(8)))  float    v8f;

namespace {
constexpr int cN   = 4096;   // tokens
constexpr int cH   = 512;    // hidden
constexpr int cK   = 64;     // categories
constexpr int cC   = 128;    // children per category
constexpr int cM   = 4;      // LSTM cells
constexpr int cIN  = 1024;   // E + H
constexpr int cG   = 4 * cH; // 2048 gates
}

// ---- workspace layout (bytes) ---------------------------------------------
// [Wlin16: K*C*H f16][Wih16: M*G*IN f16][Whh16: M*G*H f16]
// [cat_cnt: K i32][ci_cnt: M i32][pad][cat_list: K*N i32][ci_list: M*N i32]
static constexpr size_t WS_WLIN = 0;
static constexpr size_t WS_WIH  = WS_WLIN + (size_t)cK * cC * cH * 2;
static constexpr size_t WS_WHH  = WS_WIH  + (size_t)cM * cG * cIN * 2;
static constexpr size_t WS_CCNT = WS_WHH  + (size_t)cM * cG * cH * 2;
static constexpr size_t WS_MCNT = WS_CCNT + (size_t)cK * 4;
static constexpr size_t WS_CLST = ((WS_MCNT + (size_t)cM * 4 + 255) / 256) * 256;
static constexpr size_t WS_MLST = WS_CLST + (size_t)cK * cN * 4;

// ---------------------------------------------------------------------------
__global__ __launch_bounds__(256) void k_zero(int* cat_cnt, int* ci_cnt) {
    int t = threadIdx.x;
    if (t < cK) cat_cnt[t] = 0;
    if (t < cM) ci_cnt[t]  = 0;
}

__global__ __launch_bounds__(256) void k_cvt(const float* __restrict__ s,
                                             _Float16* __restrict__ d, int n4) {
    int i = blockIdx.x * blockDim.x + threadIdx.x;
    int stride = gridDim.x * blockDim.x;
    for (; i < n4; i += stride) {
        float4 v = ((const float4*)s)[i];
        _Float16* p = d + (size_t)i * 4;
        p[0] = (_Float16)v.x; p[1] = (_Float16)v.y;
        p[2] = (_Float16)v.z; p[3] = (_Float16)v.w;
    }
}

__global__ __launch_bounds__(256) void k_bin(const int* __restrict__ cat,
                                             const int* __restrict__ ci,
                                             int* cat_cnt, int* cat_list,
                                             int* ci_cnt, int* ci_list) {
    int n = blockIdx.x * blockDim.x + threadIdx.x;
    if (n >= cN) return;
    int k = cat[n];
    int p = atomicAdd(&cat_cnt[k], 1);
    cat_list[(size_t)k * cN + p] = n;
    int m = ci[n];
    p = atomicAdd(&ci_cnt[m], 1);
    ci_list[(size_t)m * cN + p] = n;
}

// ---- WMMA helpers ----------------------------------------------------------
__device__ inline v8f wmma16(v16h a, v16h b, v8f c) {
    return __builtin_amdgcn_wmma_f32_16x16x32_f16(
        /*neg_a=*/false, a, /*neg_b=*/false, b,
        /*c_mod=*/(short)0, c, /*reuse_a=*/false, /*reuse_b=*/false);
}

// A-fragment (16x32 f16, row m = lane&15) from an LDS row:
// lanes 0-15 hold K {kk+0..15}, lanes 16-31 hold K {kk+16..31}.
__device__ inline v16h a_frag(const _Float16* rowPtr, int kk, int hi) {
    v8h lo = *(const v8h*)(rowPtr + kk + 16 * hi);
    v8h hh = *(const v8h*)(rowPtr + kk + 16 * hi + 8);
    v16h a;
#pragma unroll
    for (int e = 0; e < 8; ++e) { a[e] = lo[e]; a[e + 8] = hh[e]; }
    return a;
}

// ---------------------------------------------------------------------------
// log_odds: one block per (category, 16-token tile); 8 waves -> 8 c-tiles of 16.
__global__ __launch_bounds__(256) void k_logodds(const float* __restrict__ et,
                                                 const float* __restrict__ b_lin,
                                                 const _Float16* __restrict__ Wlin16,
                                                 const int* __restrict__ cat_cnt,
                                                 const int* __restrict__ cat_list,
                                                 float* __restrict__ out_lo) {
    const int katg = blockIdx.x;
    const int tile = blockIdx.y;
    const int cnt  = cat_cnt[katg];
    if (tile * 16 >= cnt) return;

    __shared__ int toks[16];
    __shared__ __align__(16) _Float16 sA[16 * cH];   // 16 KB

    const int tid = threadIdx.x;
    if (tid < 16) {
        int idx = tile * 16 + tid;
        toks[tid] = cat_list[(size_t)katg * cN + (idx < cnt ? idx : cnt - 1)];
    }
    __syncthreads();

    // stage 16 et rows as f16 (f32 gather -> LDS)
    for (int ch = tid; ch < 16 * (cH / 4); ch += 256) {
        int row = ch >> 7;            // 128 float4 per row
        int c4  = ch & 127;
        float4 v = ((const float4*)(et + (size_t)toks[row] * cH))[c4];
        _Float16* d = &sA[row * cH + c4 * 4];
        d[0] = (_Float16)v.x; d[1] = (_Float16)v.y;
        d[2] = (_Float16)v.z; d[3] = (_Float16)v.w;
    }
    __syncthreads();

    const int wave = tid >> 5, lane = tid & 31;
    const int col = lane & 15, hi = lane >> 4;
    const int cbase = wave * 16;
    const _Float16* arow = &sA[col * cH];                       // A row = lane&15
    const _Float16* wb = Wlin16 + ((size_t)katg * cC + cbase + col) * cH; // B col = lane&15

    v8f acc = {};
    for (int kk = 0; kk < cH; kk += 32) {
        v16h a = a_frag(arow, kk, hi);
        v16h b = *(const v16h*)(wb + kk + 16 * hi);
        acc = wmma16(a, b, acc);
    }

    const float bias = b_lin[katg * cC + cbase + col];
#pragma unroll
    for (int r = 0; r < 8; ++r) {
        int m = r + 8 * hi;
        if (tile * 16 + m < cnt)
            out_lo[(size_t)toks[m] * cC + cbase + col] = acc[r] + bias;
    }
}

// ---------------------------------------------------------------------------
// LSTM: one block per (cell, 16-token tile). A = concat[x|h] (16x1536) in LDS.
// Wave w handles hh-tiles {w, w+8, w+16, w+24}; i/f/g/o accumulated together,
// pointwise LSTM fused in the register epilogue.
__global__ __launch_bounds__(256) void k_lstm(const float* __restrict__ x,
                                              const float* __restrict__ h,
                                              const float* __restrict__ c_in,
                                              const float* __restrict__ b_ih,
                                              const float* __restrict__ b_hh,
                                              const _Float16* __restrict__ Wih16,
                                              const _Float16* __restrict__ Whh16,
                                              const int* __restrict__ ci_cnt,
                                              const int* __restrict__ ci_list,
                                              float* __restrict__ out_h,
                                              float* __restrict__ out_c) {
    const int cell = blockIdx.x;
    const int tile = blockIdx.y;
    const int cnt  = ci_cnt[cell];
    if (tile * 16 >= cnt) return;

    __shared__ int toks[16];
    extern __shared__ __align__(16) _Float16 sA[];   // 16 x 1536 f16 = 48 KB

    const int tid = threadIdx.x;
    if (tid < 16) {
        int idx = tile * 16 + tid;
        toks[tid] = ci_list[(size_t)cell * cN + (idx < cnt ? idx : cnt - 1)];
    }
    __syncthreads();

    // stage concat [x(1024) | h(512)] rows as f16
    constexpr int ROW = cIN + cH;                    // 1536
    for (int ch = tid; ch < 16 * (ROW / 4); ch += 256) {
        int row = ch / (ROW / 4);
        int c4  = ch % (ROW / 4);
        float4 v;
        if (c4 < cIN / 4) v = ((const float4*)(x + (size_t)toks[row] * cIN))[c4];
        else              v = ((const float4*)(h + (size_t)toks[row] * cH))[c4 - cIN / 4];
        _Float16* d = &sA[row * ROW + c4 * 4];
        d[0] = (_Float16)v.x; d[1] = (_Float16)v.y;
        d[2] = (_Float16)v.z; d[3] = (_Float16)v.w;
    }
    __syncthreads();

    const int wave = tid >> 5, lane = tid & 31;
    const int col = lane & 15, hi = lane >> 4;
    const _Float16* arow = &sA[col * ROW];
    const _Float16* Wih = Wih16 + (size_t)cell * cG * cIN;
    const _Float16* Whh = Whh16 + (size_t)cell * cG * cH;

    for (int it = 0; it < 4; ++it) {
        const int hbase = (wave + 8 * it) * 16;      // 0..496
        const int g0 = 0 * cH + hbase + col;         // i
        const int g1 = 1 * cH + hbase + col;         // f
        const int g2 = 2 * cH + hbase + col;         // g
        const int g3 = 3 * cH + hbase + col;         // o

        v8f acc0 = {}, acc1 = {}, acc2 = {}, acc3 = {};
        for (int kk = 0; kk < cIN; kk += 32) {
            v16h a = a_frag(arow, kk, hi);
            acc0 = wmma16(a, *(const v16h*)(Wih + (size_t)g0 * cIN + kk + 16 * hi), acc0);
            acc1 = wmma16(a, *(const v16h*)(Wih + (size_t)g1 * cIN + kk + 16 * hi), acc1);
            acc2 = wmma16(a, *(const v16h*)(Wih + (size_t)g2 * cIN + kk + 16 * hi), acc2);
            acc3 = wmma16(a, *(const v16h*)(Wih + (size_t)g3 * cIN + kk + 16 * hi), acc3);
        }
        for (int kk = 0; kk < cH; kk += 32) {
            v16h a = a_frag(arow, cIN + kk, hi);
            acc0 = wmma16(a, *(const v16h*)(Whh + (size_t)g0 * cH + kk + 16 * hi), acc0);
            acc1 = wmma16(a, *(const v16h*)(Whh + (size_t)g1 * cH + kk + 16 * hi), acc1);
            acc2 = wmma16(a, *(const v16h*)(Whh + (size_t)g2 * cH + kk + 16 * hi), acc2);
            acc3 = wmma16(a, *(const v16h*)(Whh + (size_t)g3 * cH + kk + 16 * hi), acc3);
        }

        const size_t bb = (size_t)cell * cG;
        const float bi = b_ih[bb + g0] + b_hh[bb + g0];
        const float bf = b_ih[bb + g1] + b_hh[bb + g1];
        const float bg = b_ih[bb + g2] + b_hh[bb + g2];
        const float bo = b_ih[bb + g3] + b_hh[bb + g3];
        const int hcol = hbase + col;

#pragma unroll
        for (int r = 0; r < 8; ++r) {
            int m = r + 8 * hi;
            if (tile * 16 + m < cnt) {
                int tok = toks[m];
                float iv = acc0[r] + bi, fv = acc1[r] + bf;
                float gv = acc2[r] + bg, ov = acc3[r] + bo;
                float si = 1.0f / (1.0f + __expf(-iv));
                float sf = 1.0f / (1.0f + __expf(-fv));
                float so = 1.0f / (1.0f + __expf(-ov));
                float cold = c_in[(size_t)tok * cH + hcol];
                float c2 = sf * cold + si * tanhf(gv);
                float h2 = so * tanhf(c2);
                out_h[(size_t)tok * cH + hcol] = h2;
                out_c[(size_t)tok * cH + hcol] = c2;
            }
        }
    }
}

// ---------------------------------------------------------------------------
extern "C" void kernel_launch(void* const* d_in, const int* in_sizes, int n_in,
                              void* d_out, int out_size, void* d_ws, size_t ws_size,
                              hipStream_t stream) {
    const float* et    = (const float*)d_in[0];
    const float* x     = (const float*)d_in[1];
    const float* h     = (const float*)d_in[2];
    const float* c     = (const float*)d_in[3];
    const int*   cat   = (const int*)d_in[4];
    const int*   ci    = (const int*)d_in[5];
    const float* W_lin = (const float*)d_in[6];
    const float* b_lin = (const float*)d_in[7];
    const float* W_ih  = (const float*)d_in[8];
    const float* W_hh  = (const float*)d_in[9];
    const float* b_ih  = (const float*)d_in[10];
    const float* b_hh  = (const float*)d_in[11];

    float* out      = (float*)d_out;
    float* out_lo   = out;                          // [N, C]
    float* out_h2   = out + (size_t)cN * cC;        // [N, H]
    float* out_c2   = out_h2 + (size_t)cN * cH;     // [N, H]

    char* ws = (char*)d_ws;
    _Float16* Wlin16  = (_Float16*)(ws + WS_WLIN);
    _Float16* Wih16   = (_Float16*)(ws + WS_WIH);
    _Float16* Whh16   = (_Float16*)(ws + WS_WHH);
    int* cat_cnt      = (int*)(ws + WS_CCNT);
    int* ci_cnt       = (int*)(ws + WS_MCNT);
    int* cat_list     = (int*)(ws + WS_CLST);
    int* ci_list      = (int*)(ws + WS_MLST);

    // 1) re-init counters (ws is poisoned; must be done every call)
    k_zero<<<1, 256, 0, stream>>>(cat_cnt, ci_cnt);

    // 2) weight conversion f32 -> f16 (L2-resident thereafter)
    k_cvt<<<1024, 256, 0, stream>>>(W_lin, Wlin16, cK * cC * cH / 4);
    k_cvt<<<1024, 256, 0, stream>>>(W_ih,  Wih16,  cM * cG * cIN / 4);
    k_cvt<<<1024, 256, 0, stream>>>(W_hh,  Whh16,  cM * cG * cH / 4);

    // 3) route tokens by category and by child index
    k_bin<<<cN / 256, 256, 0, stream>>>(cat, ci, cat_cnt, cat_list, ci_cnt, ci_list);

    // 4) gathered per-category GEMV-as-GEMM -> log_odds
    k_logodds<<<dim3(cK, cN / 16), 256, 0, stream>>>(et, b_lin, Wlin16,
                                                     cat_cnt, cat_list, out_lo);

    // 5) routed LSTM cell GEMM + fused pointwise epilogue
    size_t lstm_shm = (size_t)16 * (cIN + cH) * sizeof(_Float16);   // 48 KB
    k_lstm<<<dim3(cM, cN / 16), 256, lstm_shm, stream>>>(x, h, c, b_ih, b_hh,
                                                         Wih16, Whh16,
                                                         ci_cnt, ci_list,
                                                         out_h2, out_c2);
}
